// SparseFactorisationConv2D_76682346103526
// MI455X (gfx1250) — compile-verified, matrix operand order checked by reading, not run on hardware
//
#include <hip/hip_runtime.h>

// ---------------------------------------------------------------------------
// Problem constants (from reference)
// ---------------------------------------------------------------------------
#define BATCH 32
#define HH    32
#define WW    32
#define CC    128
#define KHW   3
#define FF    512
#define KDIM  (KHW * KHW * CC)   // 1152
#define MDIM  (BATCH * HH * WW)  // 32768
#define D1    1152               // hidden dim of factor chain

typedef __attribute__((ext_vector_type(16))) _Float16 v16h;
typedef __attribute__((ext_vector_type(8)))  _Float16 v8h;
typedef __attribute__((ext_vector_type(8)))  float    v8f;

// ---------------------------------------------------------------------------
// Weight prep GEMM 1: t (1152x512) = (k0*m0)(1152x1152) @ (k1*m1)(1152x512)
// fp32, LDS tiled 16x16. Tiny (~1.4 GFLOP) -> plain VALU is fine.
// ---------------------------------------------------------------------------
__global__ __launch_bounds__(256)
void wprep1_kernel(const float* __restrict__ k0, const float* __restrict__ m0,
                   const float* __restrict__ k1, const float* __restrict__ m1,
                   float* __restrict__ t) {
  __shared__ float sA[16][17];
  __shared__ float sB[16][17];
  const int tx = threadIdx.x & 15;
  const int ty = threadIdx.x >> 4;
  const int row = blockIdx.y * 16 + ty;   // 0..1151
  const int col = blockIdx.x * 16 + tx;   // 0..511
  float acc = 0.f;
  for (int kt = 0; kt < D1; kt += 16) {
    const int ka = kt + tx;
    const int kb = kt + ty;
    sA[ty][tx] = k0[row * D1 + ka] * m0[row * D1 + ka];
    sB[ty][tx] = k1[kb * FF + col] * m1[kb * FF + col];
    __syncthreads();
#pragma unroll
    for (int kk = 0; kk < 16; ++kk) acc += sA[ty][kk] * sB[kk][tx];
    __syncthreads();
  }
  t[row * FF + col] = acc;
}

// ---------------------------------------------------------------------------
// Weight prep GEMM 2: wht (512x1152 f16, TRANSPOSED) = scale*(t @ (k2*m2))^T
// Storing [n][k] makes conv-side B staging pure b128 copies (no transpose).
// Row index k of the logical filter is (kh*3+kw)*128+c == the HWIO flatten.
// ---------------------------------------------------------------------------
__global__ __launch_bounds__(256)
void wprep2_kernel(const float* __restrict__ t,
                   const float* __restrict__ k2, const float* __restrict__ m2,
                   const float* __restrict__ scale,
                   _Float16* __restrict__ wht) {
  __shared__ float sA[16][17];
  __shared__ float sB[16][17];
  const int tx = threadIdx.x & 15;
  const int ty = threadIdx.x >> 4;
  const int row = blockIdx.y * 16 + ty;   // 0..1151  (k index)
  const int col = blockIdx.x * 16 + tx;   // 0..511   (n index)
  float acc = 0.f;
  for (int kt = 0; kt < FF; kt += 16) {
    const int ka = kt + tx;
    const int kb = kt + ty;
    sA[ty][tx] = t[row * FF + ka];
    sB[ty][tx] = k2[kb * FF + col] * m2[kb * FF + col];
    __syncthreads();
#pragma unroll
    for (int kk = 0; kk < 16; ++kk) acc += sA[ty][kk] * sB[kk][tx];
    __syncthreads();
  }
  wht[(size_t)col * KDIM + row] = (_Float16)(acc * scale[0]);
}

// ---------------------------------------------------------------------------
// Implicit-GEMM conv, f16 WMMA with f32 accumulate, double-buffered LDS.
//   M = 32768 pixels, N = 512 filters, K = 1152 (9 planes x 128 channels)
// Block = 256 threads = 8 wave32s, macro tile 64(M) x 128(N).
// Waves in a 2x4 grid; each wave owns a 2x2 grid of 16x16 D tiles
//   -> 4 v_wmma per K-step from 2 A + 2 B fragments (each reused twice).
// LDS rows padded to 40 halves (80 B) -> conflict-free ds_load_b128.
// ---------------------------------------------------------------------------
#define LDA 40

__global__ __launch_bounds__(256)
void conv_wmma_kernel(const float* __restrict__ x,
                      const _Float16* __restrict__ wht,   // [512][1152] f16
                      const float* __restrict__ bias,
                      float* __restrict__ out) {
  __shared__ __align__(16) _Float16 ldsA[2][64 * LDA];    // pixels x k
  __shared__ __align__(16) _Float16 ldsB[2][128 * LDA];   // filters x k

  const int tid    = threadIdx.x;
  const int m_base = blockIdx.y * 64;
  const int n_base = blockIdx.x * 128;

  const int wave = tid >> 5;
  const int lane = tid & 31;
  const int wm   = wave >> 2;        // 0..1 : 32-row M strip
  const int wn   = wave & 3;         // 0..3 : 32-col N strip
  const int row  = lane & 15;        // fragment row (M or N)
  const int hi   = lane >> 4;        // lane-half selector
  const int kfr  = hi * 8;           // per-lane K base within 32-chunk

  // A stage: 4 threads per pixel, 8 channels each (2x float4 -> v8h)
  const int p  = tid >> 2;           // pixel 0..63 within tile
  const int cq = (tid & 3) * 8;      // channel offset 0..24
  const int m  = m_base + p;
  const int bb = m >> 10;
  const int h  = (m >> 5) & 31;
  const int w  = m & 31;

  v8f acc00 = {}, acc01 = {}, acc10 = {}, acc11 = {};

  auto stage = [&](int kt, int buf) {
    const int kg  = kt * 32;
    const int kh  = kg / 384;              // 3x3 plane row
    const int rem = kg - kh * 384;
    const int kw  = rem >> 7;              // plane col
    const int c0  = rem & 127;             // channel base (32 | 128 -> uniform)
    const int ih  = h + kh - 1;
    const int iw  = w + kw - 1;

    // ---- im2col gather, fp32 -> f16 ----
    v8h a8 = {(_Float16)0, (_Float16)0, (_Float16)0, (_Float16)0,
              (_Float16)0, (_Float16)0, (_Float16)0, (_Float16)0};
    if (ih >= 0 && ih < HH && iw >= 0 && iw < WW) {
      const float* src = x + (((bb * HH + ih) * WW + iw) * CC + c0 + cq);
      const float4 f0 = *(const float4*)(src);
      const float4 f1 = *(const float4*)(src + 4);
      a8 = (v8h){(_Float16)f0.x, (_Float16)f0.y, (_Float16)f0.z, (_Float16)f0.w,
                 (_Float16)f1.x, (_Float16)f1.y, (_Float16)f1.z, (_Float16)f1.w};
    }
    *(v8h*)(&ldsA[buf][p * LDA + cq]) = a8;

    // ---- weights: straight b128 copy (pre-transposed [n][k]) ----
#pragma unroll
    for (int j = 0; j < 2; ++j) {
      const int idx = tid + j * 256;
      const int nr  = idx >> 2;            // 0..127
      const int kq  = (idx & 3) * 8;       // 0..24
      const v8h bv = *(const v8h*)(wht + (size_t)(n_base + nr) * KDIM + kg + kq);
      *(v8h*)(&ldsB[buf][nr * LDA + kq]) = bv;
    }
  };

  stage(0, 0);

  for (int kt = 0; kt < 36; ++kt) {
    __syncthreads();
    if (kt < 35) stage(kt + 1, (kt + 1) & 1);
    if (kt < 34)   // pull k+2 weight tile toward L2/L0 (global_prefetch_b8)
      __builtin_prefetch(wht + (size_t)(n_base + (tid >> 1)) * KDIM + (kt + 2) * 32, 0, 3);

    const int buf = kt & 1;
    // fragments per ISA 16-bit layout:
    //   lane<16: K {0..7,16..23}; lane>=16: K {8..15,24..31}
    union { v16h v; v8h h8[2]; } A0, A1, B0, B1;
    const _Float16* aRow = &ldsA[buf][(wm * 32 + row) * LDA + kfr];
    A0.h8[0] = *(const v8h*)(aRow);
    A0.h8[1] = *(const v8h*)(aRow + 16);
    A1.h8[0] = *(const v8h*)(aRow + 16 * LDA);
    A1.h8[1] = *(const v8h*)(aRow + 16 * LDA + 16);
    const _Float16* bRow = &ldsB[buf][(wn * 32 + row) * LDA + kfr];
    B0.h8[0] = *(const v8h*)(bRow);
    B0.h8[1] = *(const v8h*)(bRow + 16);
    B1.h8[0] = *(const v8h*)(bRow + 16 * LDA);
    B1.h8[1] = *(const v8h*)(bRow + 16 * LDA + 16);

    acc00 = __builtin_amdgcn_wmma_f32_16x16x32_f16(false, A0.v, false, B0.v,
                                                   (short)0, acc00, false, false);
    acc01 = __builtin_amdgcn_wmma_f32_16x16x32_f16(false, A0.v, false, B1.v,
                                                   (short)0, acc01, false, false);
    acc10 = __builtin_amdgcn_wmma_f32_16x16x32_f16(false, A1.v, false, B0.v,
                                                   (short)0, acc10, false, false);
    acc11 = __builtin_amdgcn_wmma_f32_16x16x32_f16(false, A1.v, false, B1.v,
                                                   (short)0, acc11, false, false);
  }

  // ---- epilogue: D layout VGPR r -> M = r + 8*hi, N = lane&15; fuse bias ----
  const int n0  = n_base + wn * 32 + row;
  const int n1  = n0 + 16;
  const float bv0 = bias[n0];
  const float bv1 = bias[n1];
  const int mo0 = m_base + wm * 32 + hi * 8;
  const int mo1 = mo0 + 16;
#pragma unroll
  for (int r = 0; r < 8; ++r) {
    out[(size_t)(mo0 + r) * FF + n0] = acc00[r] + bv0;
    out[(size_t)(mo0 + r) * FF + n1] = acc01[r] + bv1;
    out[(size_t)(mo1 + r) * FF + n0] = acc10[r] + bv0;
    out[(size_t)(mo1 + r) * FF + n1] = acc11[r] + bv1;
  }
}

// ---------------------------------------------------------------------------
// Launch
// ---------------------------------------------------------------------------
extern "C" void kernel_launch(void* const* d_in, const int* in_sizes, int n_in,
                              void* d_out, int out_size, void* d_ws, size_t ws_size,
                              hipStream_t stream) {
  const float* x     = (const float*)d_in[0];
  const float* k0    = (const float*)d_in[1];
  const float* k1    = (const float*)d_in[2];
  const float* k2    = (const float*)d_in[3];
  const float* m0    = (const float*)d_in[4];
  const float* m1    = (const float*)d_in[5];
  const float* m2    = (const float*)d_in[6];
  const float* scale = (const float*)d_in[7];
  const float* bias  = (const float*)d_in[8];
  float* out = (float*)d_out;

  // workspace: t (1152x512 f32) | wht (512x1152 f16, transposed)
  float*    t   = (float*)d_ws;
  _Float16* wht = (_Float16*)((char*)d_ws + (size_t)KDIM * FF * sizeof(float));

  dim3 gW(FF / 16, KDIM / 16);     // (32, 72)
  wprep1_kernel<<<gW, 256, 0, stream>>>(k0, m0, k1, m1, t);
  wprep2_kernel<<<gW, 256, 0, stream>>>(t, k2, m2, scale, wht);

  dim3 gC(FF / 128, MDIM / 64);    // (4, 512)
  conv_wmma_kernel<<<gC, 256, 0, stream>>>(x, wht, bias, out);
  (void)in_sizes; (void)n_in; (void)out_size; (void)ws_size;
}
